// ComplexNeuronRNN_32315333935404
// MI455X (gfx1250) — matrix operand order repaired and checked by reading, not compiled, not run on hardware
//
#include <hip/hip_runtime.h>
#include <hip/hip_bf16.h>

typedef __attribute__((ext_vector_type(16))) _Float16 v16h;
typedef __attribute__((ext_vector_type(8)))  _Float16 v8h;
typedef __attribute__((ext_vector_type(8)))  float    v8f;

#define TT 64
#define BB 64
#define EMBD 512
#define HH 512
#define HAD 2048
#define NTOK 50257
#define NTOKPAD 50304   // padded vocab rows: 393 * 128

__device__ __forceinline__ v16h combine16(v8h lo, v8h hi) {
  v16h r;
#pragma unroll
  for (int i = 0; i < 8; ++i) { r[i] = lo[i]; r[i + 8] = hi[i]; }
  return r;
}

// ---------------------------------------------------------------------------
// WMMA fragment loaders (CDNA5 wave32, V_WMMA_F32_16X16X32_F16)
// A 16x32 f16 layout: lane L holds row m=L%16; halves j: K = (L/16)*16 + j
// B 32x16 f16 layout: lane L holds col n=L%16; halves j: K = (L/16)*16 + j
// B fragments load from pre-transposed weights W^T stored row-major [N, K].
// ---------------------------------------------------------------------------
__device__ __forceinline__ v16h load_a_frag(const _Float16* __restrict__ base,
                                            int ld, int m0, int k0, int lane) {
  const _Float16* p = base + (size_t)(m0 + (lane & 15)) * ld + (k0 + ((lane >> 4) << 4));
  return combine16(*(const v8h*)p, *(const v8h*)(p + 8));
}

__device__ __forceinline__ v16h load_b_frag(const _Float16* __restrict__ bt,
                                            int ld, int n0, int k0, int lane) {
  const _Float16* p = bt + (size_t)(n0 + (lane & 15)) * ld + (k0 + ((lane >> 4) << 4));
  return *(const v16h*)p;   // 32 contiguous bytes, 32B aligned
}

__device__ __forceinline__ v8f wmma_f16(v16h a, v16h b, v8f c) {
  return __builtin_amdgcn_wmma_f32_16x16x32_f16(false, a, false, b, (short)0, c, false, false);
}

// Async global -> LDS copy of one 128 x 64-half chunk of B^T (CDNA5 ASYNCcnt path)
__device__ __forceinline__ void async_copy_B(const _Float16* __restrict__ BT,
                                             _Float16* lds_dst, int nblk, int k0,
                                             int tid) {
  for (int c = tid; c < 1024; c += 256) {          // 1024 x 16B chunks = 32KB/2
    int row = c >> 3, cp = c & 7;
    unsigned da = (unsigned)(size_t)(lds_dst + row * 64 + cp * 8);
    unsigned long long sa =
        (unsigned long long)(size_t)(BT + (size_t)(nblk + row) * 512 + k0 + cp * 8);
    asm volatile("global_load_async_to_lds_b128 %0, %1, off"
                 :: "v"(da), "v"(sa) : "memory");
  }
}

__device__ __forceinline__ void wait_asynccnt0() {
  asm volatile("s_wait_asynccnt 0" ::: "memory");
}

// ---------------------------------------------------------------------------
// One-time prep: f32 [K,N] -> f16 [Npad,K] transpose (LDS tiled, pad rows = 0)
// ---------------------------------------------------------------------------
__global__ __launch_bounds__(256)
void cnr_transpose_cvt(const float* __restrict__ in, _Float16* __restrict__ out,
                       int K, int N) {
  __shared__ float tile[32][33];
  const int n0 = blockIdx.x * 32, k0 = blockIdx.y * 32;
  const int tx = threadIdx.x, ty = threadIdx.y;          // blockDim = (32,8)
#pragma unroll
  for (int i = 0; i < 32; i += 8) {
    int n = n0 + tx;
    tile[ty + i][tx] = (n < N) ? in[(size_t)(k0 + ty + i) * N + n] : 0.0f;
  }
  __syncthreads();
#pragma unroll
  for (int i = 0; i < 32; i += 8) {
    out[(size_t)(n0 + ty + i) * K + (k0 + tx)] = (_Float16)tile[tx][ty + i];
  }
}

// ---------------------------------------------------------------------------
// Embedding gather -> f16 [T,B,512]
// ---------------------------------------------------------------------------
__global__ __launch_bounds__(256)
void cnr_embed(const int* __restrict__ x, const float* __restrict__ enc,
               _Float16* __restrict__ out) {
  int idx = blockIdx.x * 256 + threadIdx.x;              // < 64*64*512
  int tb = idx >> 9, e = idx & 511;
  out[idx] = (_Float16)enc[(size_t)x[tb] * EMBD + e];
}

__global__ __launch_bounds__(256)
void cnr_init(_Float16* __restrict__ h0, _Float16* __restrict__ h1,
              float* __restrict__ loss) {
  int i = blockIdx.x * 256 + threadIdx.x;
  if (i < BB * HH) { h0[i] = (_Float16)0.f; h1[i] = (_Float16)0.f; }
  if (i == 0) *loss = 0.f;
}

// ---------------------------------------------------------------------------
// One RNN cell step (one layer), grid = 16 WGs, each owns a 64x128 tile of
// pre = X@Wx + H@Wh (+ biases), full K=512 in-workgroup, then the per-neuron
// 4->16->16->1 MLP on that tile. 8 waves: 4 M-tiles x 2 N-groups.
// Latency-bound (serial over t) -> no LDS staging/barriers in the GEMM.
// ---------------------------------------------------------------------------
__global__ __launch_bounds__(256)
void cnr_rnn_step(const _Float16* __restrict__ X,      // [64,512] f16
                  const _Float16* __restrict__ Hprev,  // [64,512] f16
                  const _Float16* __restrict__ WxT,    // [2048,512] f16
                  const _Float16* __restrict__ WhT,    // [2048,512] f16
                  const float* __restrict__ bx, const float* __restrict__ bh,
                  const float* __restrict__ W1, const float* __restrict__ b1,
                  const float* __restrict__ W2, const float* __restrict__ b2,
                  const float* __restrict__ W3, const float* __restrict__ b3,
                  _Float16* __restrict__ hout,         // [64,512] f16
                  _Float16* __restrict__ hout2,        // optional f16 copy (h1all)
                  float* __restrict__ houtf32,         // optional f32 (final h out)
                  float* __restrict__ preout) {        // optional f32 in2cells out
  __shared__ float pre[64 * 128];
  __shared__ float mW1[64], mB1[16], mW2[256], mB2[16], mW3[16], mB3[1];

  const int tid = threadIdx.x;
  const int lane = tid & 31;
  const int wave = tid >> 5;
  const int n0 = blockIdx.x * 128;       // tile never straddles a cell type
  const int cell = n0 >> 9;              // n0 / 512
  const int m0 = (wave & 3) * 16;
  const int ng = (wave >> 2) * 64;

  // stage tiny-MLP params for this cell type into LDS
  if (tid < 64) mW1[tid] = W1[cell * 64 + tid];
  if (tid < 16) { mB1[tid] = b1[cell * 16 + tid];
                  mB2[tid] = b2[cell * 16 + tid];
                  mW3[tid] = W3[cell * 16 + tid]; }
  if (tid == 0) mB3[0] = b3[cell];
  mW2[tid] = W2[cell * 256 + tid];

  v8f acc[4];
#pragma unroll
  for (int i = 0; i < 4; ++i)
#pragma unroll
    for (int j = 0; j < 8; ++j) acc[i][j] = 0.0f;

#pragma unroll
  for (int s = 0; s < 2; ++s) {
    const _Float16* Asrc = s ? Hprev : X;
    const _Float16* Bw   = s ? WhT   : WxT;
    for (int k0 = 0; k0 < 512; k0 += 32) {
      v16h a = load_a_frag(Asrc, 512, m0, k0, lane);
#pragma unroll
      for (int nt = 0; nt < 4; ++nt) {
        v16h b = load_b_frag(Bw, 512, n0 + ng + nt * 16, k0, lane);
        acc[nt] = wmma_f16(a, b, acc[nt]);
      }
    }
  }

  // C/D layout: VGPR r, lane L -> row m0 + r + 8*(L/16), col (L&15)
  {
    const int rbase = m0 + ((lane >> 4) << 3);
    const int cl = lane & 15;
#pragma unroll
    for (int nt = 0; nt < 4; ++nt) {
      int coll = ng + nt * 16 + cl;
      float bias = bx[n0 + coll] + bh[n0 + coll];
#pragma unroll
      for (int r = 0; r < 8; ++r)
        pre[(rbase + r) * 128 + coll] = acc[nt][r] + bias;
    }
  }
  __syncthreads();

  if (preout) {   // in2cells of the last timestep: just `pre` reshaped
    for (int i = tid; i < 64 * 128; i += 256) {
      int b = i >> 7, c = i & 127;
      preout[b * HAD + n0 + c] = pre[i];
    }
  }

  // per-neuron MLP: 64 rows x 32 neurons in tile
  for (int i = tid; i < 64 * 32; i += 256) {
    int b = i >> 5, g = i & 31;
    const float* ar = &pre[b * 128 + g * 4];
    float h1v[16], h2v[16];
#pragma unroll
    for (int j = 0; j < 16; ++j) {
      float s = mB1[j];
#pragma unroll
      for (int a2 = 0; a2 < 4; ++a2) s += ar[a2] * mW1[a2 * 16 + j];
      h1v[j] = s > 0.f ? s : 0.f;
    }
#pragma unroll
    for (int j = 0; j < 16; ++j) {
      float s = mB2[j];
#pragma unroll
      for (int k = 0; k < 16; ++k) s += h1v[k] * mW2[k * 16 + j];
      h2v[j] = s > 0.f ? s : 0.f;
    }
    float o = mB3[0];
#pragma unroll
    for (int j = 0; j < 16; ++j) o += h2v[j] * mW3[j];

    int hidx = (n0 >> 2) + g;
    _Float16 oh = (_Float16)o;
    hout[b * HH + hidx] = oh;
    if (hout2)   hout2[b * HH + hidx] = oh;
    if (houtf32) houtf32[b * HH + hidx] = o;
  }
}

// ---------------------------------------------------------------------------
// Decoder GEMM: [4096,512] f16 @ dec^T [50304,512] f16 -> logits f32.
// dec^T (51.5 MB) is L2-resident; each WG pulls its B tile into LDS exactly
// once per 64-K chunk via async global->LDS (double-buffered, ASYNCcnt),
// overlapping the copy with WMMA on the previous chunk.
// ---------------------------------------------------------------------------
__global__ __launch_bounds__(256)
void cnr_decoder(const _Float16* __restrict__ Aall,   // [4096,512]
                 const _Float16* __restrict__ BT,     // [50304,512]
                 const float* __restrict__ bias,      // [50257]
                 float* __restrict__ out) {           // [4096,50257]
  __shared__ __align__(32) _Float16 ldsB[2][128 * 64];   // 2 x 16 KB
  const int tid = threadIdx.x, lane = tid & 31, wave = tid >> 5;
  const int nblk = blockIdx.x * 128;
  const int m0 = blockIdx.y * 64 + (wave & 3) * 16;
  const int ng = (wave >> 2) * 64;

  v8f acc[4];
#pragma unroll
  for (int i = 0; i < 4; ++i)
#pragma unroll
    for (int j = 0; j < 8; ++j) acc[i][j] = 0.0f;

  async_copy_B(BT, &ldsB[0][0], nblk, 0, tid);

  for (int k0 = 0; k0 < 512; k0 += 64) {
    const int buf = (k0 >> 6) & 1;
    wait_asynccnt0();        // my chunk of buf landed in LDS
    __syncthreads();         // everyone's chunk landed; prev buf free to reuse
    if (k0 + 64 < 512)
      async_copy_B(BT, &ldsB[buf ^ 1][0], nblk, k0 + 64, tid);

#pragma unroll
    for (int ks = 0; ks < 64; ks += 32) {
      v16h a = load_a_frag(Aall, 512, m0, k0 + ks, lane);
#pragma unroll
      for (int nt = 0; nt < 4; ++nt) {
        const _Float16* p =
            &ldsB[buf][(ng + nt * 16 + (lane & 15)) * 64 + ks + ((lane >> 4) << 4)];
        v16h b = combine16(*(const v8h*)p, *(const v8h*)(p + 8));
        acc[nt] = wmma_f16(a, b, acc[nt]);
      }
    }
  }

  const int rbase = m0 + ((lane >> 4) << 3);
  const int cl = lane & 15;
#pragma unroll
  for (int nt = 0; nt < 4; ++nt) {
    int col = nblk + ng + nt * 16 + cl;
    if (col < NTOK) {
      float bs = bias[col];
#pragma unroll
      for (int r = 0; r < 8; ++r)
        out[(size_t)(rbase + r) * NTOK + col] = acc[nt][r] + bs;
    }
  }
}

// ---------------------------------------------------------------------------
// Online softmax row statistic: rowstat = max + log(sum exp(x - max)), 1 pass
// ---------------------------------------------------------------------------
__global__ __launch_bounds__(256)
void cnr_rowstat(const float* __restrict__ logits, float* __restrict__ rowstat) {
  __shared__ float sm[256], ss[256];
  const int row = blockIdx.x;
  const float* p = logits + (size_t)row * NTOK;
  float m = -3.0e38f, s = 0.f;
  for (int j = threadIdx.x; j < NTOK; j += 256) {
    float x = p[j];
    if (x > m) { s = s * __expf(m - x) + 1.f; m = x; }
    else        s += __expf(x - m);
  }
  sm[threadIdx.x] = m; ss[threadIdx.x] = s;
  __syncthreads();
  for (int off = 128; off; off >>= 1) {
    if (threadIdx.x < off) {
      float m2 = sm[threadIdx.x + off], s2 = ss[threadIdx.x + off];
      float M = fmaxf(sm[threadIdx.x], m2);
      ss[threadIdx.x] = ss[threadIdx.x] * __expf(sm[threadIdx.x] - M) + s2 * __expf(m2 - M);
      sm[threadIdx.x] = M;
    }
    __syncthreads();
  }
  if (threadIdx.x == 0) rowstat[row] = sm[0] + __logf(ss[0]);
}

__global__ __launch_bounds__(256)
void cnr_loss(const float* __restrict__ rowstat, const float* __restrict__ logits,
              const int* __restrict__ labels, float* __restrict__ loss) {
  int i = blockIdx.x * 256 + threadIdx.x;
  if (i < TT * BB) {
    float term = rowstat[i] - logits[(size_t)i * NTOK + labels[i]];
    atomicAdd(loss, term * (1.0f / (TT * BB)));
  }
}

// ---------------------------------------------------------------------------
extern "C" void kernel_launch(void* const* d_in, const int* in_sizes, int n_in,
                              void* d_out, int out_size, void* d_ws, size_t ws_size,
                              hipStream_t stream) {
  (void)in_sizes; (void)n_in; (void)out_size; (void)ws_size;
  const int*   x     = (const int*)d_in[0];
  const int*   labels= (const int*)d_in[1];
  const float* enc   = (const float*)d_in[2];
  const float* Wih0  = (const float*)d_in[3];
  const float* bih0  = (const float*)d_in[4];
  const float* Whh0  = (const float*)d_in[5];
  const float* bhh0  = (const float*)d_in[6];
  const float* Wih1  = (const float*)d_in[7];
  const float* bih1  = (const float*)d_in[8];
  const float* Whh1  = (const float*)d_in[9];
  const float* bhh1  = (const float*)d_in[10];
  const float* W1    = (const float*)d_in[11];
  const float* b1    = (const float*)d_in[12];
  const float* W2    = (const float*)d_in[13];
  const float* b2    = (const float*)d_in[14];
  const float* W3    = (const float*)d_in[15];
  const float* b3    = (const float*)d_in[16];
  const float* dec_w = (const float*)d_in[17];
  const float* dec_b = (const float*)d_in[18];

  // output layout: logits | h0f | h1f | loss | in0 | in1
  float* out_logits = (float*)d_out;
  float* out_h0f  = out_logits + (size_t)TT * BB * NTOK;
  float* out_h1f  = out_h0f + BB * HH;
  float* out_loss = out_h1f + BB * HH;
  float* out_in0  = out_loss + 1;
  float* out_in1  = out_in0 + BB * HAD;

  // workspace carve (all f16 block sizes are multiples of 16 -> 32B aligned)
  _Float16* w16 = (_Float16*)d_ws;
  size_t o = 0;
  _Float16* wtih0 = w16 + o; o += (size_t)HAD * 512;
  _Float16* wthh0 = w16 + o; o += (size_t)HAD * 512;
  _Float16* wtih1 = w16 + o; o += (size_t)HAD * 512;
  _Float16* wthh1 = w16 + o; o += (size_t)HAD * 512;
  _Float16* dect  = w16 + o; o += (size_t)NTOKPAD * 512;
  _Float16* emb16 = w16 + o; o += (size_t)TT * BB * 512;
  _Float16* h1all = w16 + o; o += (size_t)TT * BB * 512;
  _Float16* h0buf = w16 + o; o += 2 * (size_t)BB * HH;
  _Float16* h1buf = w16 + o; o += 2 * (size_t)BB * HH;
  float* rowstat  = (float*)(w16 + o);   // 4096 floats

  const dim3 tb(32, 8);
  cnr_transpose_cvt<<<dim3(HAD / 32, 16), tb, 0, stream>>>(Wih0, wtih0, 512, HAD);
  cnr_transpose_cvt<<<dim3(HAD / 32, 16), tb, 0, stream>>>(Whh0, wthh0, 512, HAD);
  cnr_transpose_cvt<<<dim3(HAD / 32, 16), tb, 0, stream>>>(Wih1, wtih1, 512, HAD);
  cnr_transpose_cvt<<<dim3(HAD / 32, 16), tb, 0, stream>>>(Whh1, wthh1, 512, HAD);
  cnr_transpose_cvt<<<dim3(NTOKPAD / 32, 16), tb, 0, stream>>>(dec_w, dect, 512, NTOK);
  cnr_embed<<<(TT * BB * 512) / 256, 256, 0, stream>>>(x, enc, emb16);
  cnr_init<<<(BB * HH + 255) / 256, 256, 0, stream>>>(h0buf, h1buf, out_loss);

  for (int t = 0; t < TT; ++t) {
    const int cur = t & 1, nxt = (t + 1) & 1;
    const bool last = (t == TT - 1);
    cnr_rnn_step<<<16, 256, 0, stream>>>(
        emb16 + (size_t)t * BB * 512, h0buf + (size_t)cur * BB * HH,
        wtih0, wthh0, bih0, bhh0, W1, b1, W2, b2, W3, b3,
        h0buf + (size_t)nxt * BB * HH, (_Float16*)nullptr,
        last ? out_h0f : nullptr, last ? out_in0 : nullptr);
    cnr_rnn_step<<<16, 256, 0, stream>>>(
        h0buf + (size_t)nxt * BB * HH, h1buf + (size_t)cur * BB * HH,
        wtih1, wthh1, bih1, bhh1, W1, b1, W2, b2, W3, b3,
        h1buf + (size_t)nxt * BB * HH, h1all + (size_t)t * BB * 512,
        last ? out_h1f : nullptr, last ? out_in1 : nullptr);
  }

  cnr_decoder<<<dim3(NTOKPAD / 128, (TT * BB) / 64), 256, 0, stream>>>(
      h1all, dect, dec_b, out_logits);
  cnr_rowstat<<<TT * BB, 256, 0, stream>>>(out_logits, rowstat);
  cnr_loss<<<(TT * BB + 255) / 256, 256, 0, stream>>>(rowstat, out_logits, labels, out_loss);
}
